// GATLayer_21964462752233
// MI455X (gfx1250) — matrix operand order repaired and checked by reading, not compiled, not run on hardware
//
#include <hip/hip_runtime.h>
#include <math.h>

// Problem constants (match reference setup_inputs)
#define NN   50000
#define EE   800000
#define F    128          // feature width everywhere (IN_CH = H1*HID = OUT = 128)

typedef __attribute__((ext_vector_type(16))) _Float16 v16h;
typedef __attribute__((ext_vector_type(8)))  float    v8f;

// ---- order-preserving float<->uint encoding for deterministic atomic max ----
__device__ __forceinline__ unsigned enc_f32(float f) {
  unsigned u = __float_as_uint(f);
  return (u & 0x80000000u) ? ~u : (u | 0x80000000u);
}
__device__ __forceinline__ float dec_f32(unsigned u) {
  return (u & 0x80000000u) ? __uint_as_float(u & 0x7FFFFFFFu) : __uint_as_float(~u);
}

// ---------------------------------------------------------------------------
// Self-loop edge attr = mean of incoming edge attrs per dst (0 if none)
// ---------------------------------------------------------------------------
__global__ void count_sum_kernel(const int* __restrict__ dst,
                                 const float* __restrict__ ea,
                                 float* __restrict__ cnt, float* __restrict__ sum) {
  int e = blockIdx.x * blockDim.x + threadIdx.x;
  if (e < EE) {
    int d = dst[e];
    atomicAdd(&cnt[d], 1.0f);
    atomicAdd(&sum[d], ea[e]);
  }
}
__global__ void mean_fin_kernel(float* __restrict__ sum, const float* __restrict__ cnt) {
  int i = blockIdx.x * blockDim.x + threadIdx.x;
  if (i < NN) sum[i] = sum[i] / fmaxf(cnt[i], 1.0f);
}

// ce[h] = dot(We[h*C : h*C+C], a_edge[h, :])  -> alpha_e(edge) = ea * ce[h]
__global__ void ce_kernel(const float* __restrict__ We, const float* __restrict__ ae,
                          int H, int C, float* __restrict__ ce) {
  int h = threadIdx.x;
  if (h < H) {
    float s = 0.f;
    for (int c = 0; c < C; ++c) s += We[h * C + c] * ae[h * C + c];
    ce[h] = s;
  }
}

// ---------------------------------------------------------------------------
// WMMA GEMM:  O[N,128] = f16(X[N,128]) x f16(W[128,128]), fp32 accumulate.
// Block = 256 threads = 8 waves; blockIdx.x = row-tile (16 rows); wave id = col-tile.
// A frag (16x32 f16): lane M=lane%16, g=lane/16; e0..7 <- K=kk+g*8.., e8..15 <- K=kk+16+g*8..
// B frag (32x16 f16): lane col N=lane%16; e0..7 <- K=kk+g*8.., e8..15 <- K=kk+16+g*8.. (as A)   (column-major gather)
// C/D  : col = lane%16, row = g*8 + v
// ---------------------------------------------------------------------------
__global__ __launch_bounds__(256)
void gemm128_f16_wmma(const float* __restrict__ X, const float* __restrict__ W,
                      float* __restrict__ O) {
  const int tm   = blockIdx.x;          // 0..3124  (50000/16)
  const int tn   = threadIdx.x >> 5;    // 0..7     (128/16)
  const int lane = threadIdx.x & 31;
  const int g    = lane >> 4;
  const int r    = lane & 15;
  const size_t arow = (size_t)(tm * 16 + r) * F;
  const int col  = tn * 16 + r;

  v8f acc = {};
#pragma unroll
  for (int kk = 0; kk < F; kk += 32) {
    v16h a, b;
    const float4* pa0 = (const float4*)(X + arow + kk + g * 8);
    const float4* pa1 = (const float4*)(X + arow + kk + 16 + g * 8);
    float4 x0 = pa0[0], x1 = pa0[1], x2 = pa1[0], x3 = pa1[1];
    a[0]  = (_Float16)x0.x; a[1]  = (_Float16)x0.y; a[2]  = (_Float16)x0.z; a[3]  = (_Float16)x0.w;
    a[4]  = (_Float16)x1.x; a[5]  = (_Float16)x1.y; a[6]  = (_Float16)x1.z; a[7]  = (_Float16)x1.w;
    a[8]  = (_Float16)x2.x; a[9]  = (_Float16)x2.y; a[10] = (_Float16)x2.z; a[11] = (_Float16)x2.w;
    a[12] = (_Float16)x3.x; a[13] = (_Float16)x3.y; a[14] = (_Float16)x3.z; a[15] = (_Float16)x3.w;
    const int kb = kk + g * 8;
#pragma unroll
    for (int e = 0; e < 16; ++e)
      b[e] = (_Float16)W[(size_t)(kb + (e & 7) + ((e >> 3) << 4)) * F + col];
    acc = __builtin_amdgcn_wmma_f32_16x16x32_f16(false, a, false, b,
                                                 (short)0, acc, false, false);
  }
#pragma unroll
  for (int v = 0; v < 8; ++v)
    O[(size_t)(tm * 16 + g * 8 + v) * F + col] = acc[v];
}

// per-node attention projections: as[n,h] = h.a_src, ad[n,h] = h.a_dst
__global__ void alpha_node_kernel(const float* __restrict__ hf,
                                  const float* __restrict__ a_src,
                                  const float* __restrict__ a_dst,
                                  int H, int C, float* __restrict__ as,
                                  float* __restrict__ ad) {
  int t = blockIdx.x * blockDim.x + threadIdx.x;
  if (t >= NN * H) return;
  int node = t / H, h = t % H;
  const float* hp = hf + (size_t)node * F + h * C;
  float s1 = 0.f, s2 = 0.f;
  for (int c = 0; c < C; c += 4) {
    float4 hv = *(const float4*)(hp + c);
    float4 av = *(const float4*)(a_src + h * C + c);
    float4 dv = *(const float4*)(a_dst + h * C + c);
    s1 += hv.x * av.x + hv.y * av.y + hv.z * av.z + hv.w * av.w;
    s2 += hv.x * dv.x + hv.y * dv.y + hv.z * dv.z + hv.w * dv.w;
  }
  as[t] = s1;
  ad[t] = s2;
}

// pass A: alpha = leaky_relu(as[src]+ad[dst]+ea*ce), store + segment max (encoded)
__global__ void edge_alpha_kernel(int M, const int* __restrict__ src,
                                  const int* __restrict__ dst,
                                  const float* __restrict__ ea,
                                  const float* __restrict__ mean_ea,
                                  const float* __restrict__ as,
                                  const float* __restrict__ ad,
                                  const float* __restrict__ ce, int H,
                                  float* __restrict__ abuf,
                                  unsigned* __restrict__ amax) {
  int t = blockIdx.x * blockDim.x + threadIdx.x;
  if (t >= M * H) return;
  int e = t / H, h = t % H;
  int s, d; float w;
  if (e < EE) { s = src[e]; d = dst[e]; w = ea[e]; }
  else        { s = d = e - EE; w = mean_ea[s]; }
  float a = as[(size_t)s * H + h] + ad[(size_t)d * H + h] + w * ce[h];
  a = (a > 0.f) ? a : 0.2f * a;           // leaky_relu, slope 0.2
  abuf[t] = a;
  atomicMax(&amax[(size_t)d * H + h], enc_f32(a));
}

// pass B: ex = exp(alpha - amax[dst]); store; segment-sum denom
__global__ void edge_exp_kernel(int M, const int* __restrict__ dst, int H,
                                const unsigned* __restrict__ amax,
                                float* __restrict__ abuf,
                                float* __restrict__ denom) {
  int t = blockIdx.x * blockDim.x + threadIdx.x;
  if (t >= M * H) return;
  int e = t / H, h = t % H;
  int d = (e < EE) ? dst[e] : e - EE;
  float ex = expf(abuf[t] - dec_f32(amax[(size_t)d * H + h]));
  abuf[t] = ex;
  atomicAdd(&denom[(size_t)d * H + h], ex);
}

// pass C: out[dst] += (ex/denom) * h[src]   — one wave32 per edge, float4 per lane
__global__ __launch_bounds__(256)
void aggregate_kernel(int M, const int* __restrict__ src,
                      const int* __restrict__ dst,
                      const float* __restrict__ hf,
                      const float* __restrict__ exb,
                      const float* __restrict__ denom,
                      int H, int C, float* __restrict__ out) {
  int e = blockIdx.x * (blockDim.x >> 5) + (threadIdx.x >> 5);
  if (e >= M) return;
  int lane = threadIdx.x & 31;
  int s, d;
  if (e < EE) { s = src[e]; d = dst[e]; }
  else        { s = d = e - EE; }
  int ch   = lane * 4;          // 4 channels per lane, 32 lanes = 128
  int head = ch / C;            // constant across the float4 (C is 64 or 128)
  float att = exb[(size_t)e * H + head] /
              (denom[(size_t)d * H + head] + 1e-16f);
  float4 hv = *(const float4*)(hf + (size_t)s * F + ch);
  float* o = out + (size_t)d * F + ch;
  atomicAdd(o + 0, hv.x * att);
  atomicAdd(o + 1, hv.y * att);
  atomicAdd(o + 2, hv.z * att);
  atomicAdd(o + 3, hv.w * att);
}

__global__ void bias_elu_kernel(const float* __restrict__ b, float* __restrict__ x) {
  int t = blockIdx.x * blockDim.x + threadIdx.x;
  if (t < NN * F) {
    float v = x[t] + b[t & (F - 1)];
    x[t] = (v > 0.f) ? v : (expf(v) - 1.0f);
  }
}
__global__ void bias_add_kernel(const float* __restrict__ b, float* __restrict__ x) {
  int t = blockIdx.x * blockDim.x + threadIdx.x;
  if (t < NN * F) x[t] += b[t & (F - 1)];
}

// ---------------------------------------------------------------------------
extern "C" void kernel_launch(void* const* d_in, const int* in_sizes, int n_in,
                              void* d_out, int out_size, void* d_ws, size_t ws_size,
                              hipStream_t stream) {
  const float* x    = (const float*)d_in[0];
  const int*   ei   = (const int*)  d_in[1];
  const float* ew   = (const float*)d_in[2];
  const float* W1   = (const float*)d_in[3];
  const float* We1  = (const float*)d_in[4];
  const float* as1  = (const float*)d_in[5];
  const float* ad1  = (const float*)d_in[6];
  const float* ae1  = (const float*)d_in[7];
  const float* b1   = (const float*)d_in[8];
  const float* W2   = (const float*)d_in[9];
  const float* We2  = (const float*)d_in[10];
  const float* as2  = (const float*)d_in[11];
  const float* ad2  = (const float*)d_in[12];
  const float* ae2  = (const float*)d_in[13];
  const float* b2   = (const float*)d_in[14];
  const int* src = ei;
  const int* dst = ei + EE;
  float* out = (float*)d_out;

  const int M = EE + NN;     // edges incl. self loops

  // ---- workspace carve-up (256B aligned) ----
  char* w = (char*)d_ws;
  size_t o = 0;
  auto alloc = [&](size_t bytes) { size_t r = o; o = (o + bytes + 255) & ~(size_t)255; return r; };
  float*    h_buf  = (float*)   (w + alloc((size_t)NN * F * 4));   // projection
  float*    x2_buf = (float*)   (w + alloc((size_t)NN * F * 4));   // layer1 out / layer2 in
  float*    as_buf = (float*)   (w + alloc((size_t)NN * 2 * 4));
  float*    ad_buf = (float*)   (w + alloc((size_t)NN * 2 * 4));
  unsigned* amax   = (unsigned*)(w + alloc((size_t)NN * 2 * 4));
  float*    denom  = (float*)   (w + alloc((size_t)NN * 2 * 4));
  float*    abuf   = (float*)   (w + alloc((size_t)M  * 2 * 4));   // alpha then ex
  float*    mean_m = (float*)   (w + alloc((size_t)NN * 4));       // sum -> mean
  float*    cnt_m  = (float*)   (w + alloc((size_t)NN * 4));
  float*    ce_buf = (float*)   (w + alloc(256));
  (void)ws_size; (void)in_sizes; (void)n_in; (void)out_size;

  const int T = 256;
  auto g1 = [](long n, int t) { return dim3((unsigned)((n + t - 1) / t)); };

  // ---- self-loop mean edge attr (shared by both layers) ----
  hipMemsetAsync(cnt_m,  0, (size_t)NN * 4, stream);
  hipMemsetAsync(mean_m, 0, (size_t)NN * 4, stream);
  count_sum_kernel<<<g1(EE, T), T, 0, stream>>>(dst, ew, cnt_m, mean_m);
  mean_fin_kernel<<<g1(NN, T), T, 0, stream>>>(mean_m, cnt_m);

  // =================== Layer 1: heads=2, C=64, concat ===================
  {
    const int H = 2, C = 64;
    ce_kernel<<<1, H, 0, stream>>>(We1, ae1, H, C, ce_buf);
    gemm128_f16_wmma<<<NN / 16, 256, 0, stream>>>(x, W1, h_buf);
    alpha_node_kernel<<<g1((long)NN * H, T), T, 0, stream>>>(h_buf, as1, ad1, H, C, as_buf, ad_buf);
    hipMemsetAsync(amax,  0, (size_t)NN * H * 4, stream);
    hipMemsetAsync(denom, 0, (size_t)NN * H * 4, stream);
    edge_alpha_kernel<<<g1((long)M * H, T), T, 0, stream>>>(M, src, dst, ew, mean_m,
                                                            as_buf, ad_buf, ce_buf, H, abuf, amax);
    edge_exp_kernel<<<g1((long)M * H, T), T, 0, stream>>>(M, dst, H, amax, abuf, denom);
    hipMemsetAsync(x2_buf, 0, (size_t)NN * F * 4, stream);
    aggregate_kernel<<<g1(M, 8), 256, 0, stream>>>(M, src, dst, h_buf, abuf, denom, H, C, x2_buf);
    bias_elu_kernel<<<g1((long)NN * F, T), T, 0, stream>>>(b1, x2_buf);
  }

  // =================== Layer 2: heads=1, C=128, mean ====================
  {
    const int H = 1, C = 128;
    ce_kernel<<<1, H, 0, stream>>>(We2, ae2, H, C, ce_buf);
    gemm128_f16_wmma<<<NN / 16, 256, 0, stream>>>(x2_buf, W2, h_buf);
    alpha_node_kernel<<<g1((long)NN * H, T), T, 0, stream>>>(h_buf, as2, ad2, H, C, as_buf, ad_buf);
    hipMemsetAsync(amax,  0, (size_t)NN * H * 4, stream);
    hipMemsetAsync(denom, 0, (size_t)NN * H * 4, stream);
    edge_alpha_kernel<<<g1((long)M * H, T), T, 0, stream>>>(M, src, dst, ew, mean_m,
                                                            as_buf, ad_buf, ce_buf, H, abuf, amax);
    edge_exp_kernel<<<g1((long)M * H, T), T, 0, stream>>>(M, dst, H, amax, abuf, denom);
    hipMemsetAsync(out, 0, (size_t)NN * F * 4, stream);
    aggregate_kernel<<<g1(M, 8), 256, 0, stream>>>(M, src, dst, h_buf, abuf, denom, H, C, out);
    bias_add_kernel<<<g1((long)NN * F, T), T, 0, stream>>>(b2, out);
  }
}
